// RWKV6TimeMixB2_10359461117971
// MI455X (gfx1250) — matrix operand order repaired and checked by reading, not compiled, not run on hardware
//
#include <hip/hip_runtime.h>
#include <math.h>

#define B_    4
#define T_    2048
#define C_    2048
#define H_    32
#define N_    64
#define DMIX_ 32
#define DDEC_ 64
#define M_    (B_*T_)

typedef __attribute__((ext_vector_type(16))) __bf16 v16bf;
typedef __attribute__((ext_vector_type(8)))  float  v8f;

__device__ __forceinline__ unsigned short f32_to_bf16(float f) {
    unsigned int u = __float_as_uint(f);
    unsigned int r = u + 0x7FFFu + ((u >> 16) & 1u);   // round-to-nearest-even
    return (unsigned short)(r >> 16);
}

// gfx1250 async memory->LDS copy (CDNA5 ISA).
// LDS operand is the 32-bit LDS byte address = low 32 bits of the flat address
// (ISA 10.2: LDS aperture maps addr[31:0] directly).
__device__ __forceinline__ void async_load_b128(const void* g, void* l) {
    unsigned long long ga = (unsigned long long)(uintptr_t)g;
    unsigned int       la = (unsigned int)(uintptr_t)l;
    asm volatile("global_load_async_to_lds_b128 %0, %1, off"
                 :: "v"(la), "v"(ga) : "memory");
}
__device__ __forceinline__ void wait_async0() {
    asm volatile("s_wait_asynccnt 0x0" ::: "memory");
}

// ---------------------------------------------------------------------------
// f32 [R,Cc] -> bf16 transposed [Cc,R]
// ---------------------------------------------------------------------------
__global__ void cvt_transpose_bf16_kernel(const float* __restrict__ in,
                                          unsigned short* __restrict__ out,
                                          int R, int Cc) {
    long idx = (long)blockIdx.x * blockDim.x + threadIdx.x;
    long total = (long)R * Cc;
    if (idx < total) {
        int r = (int)(idx / Cc), c = (int)(idx % Cc);
        out[(size_t)c * R + r] = f32_to_bf16(in[idx]);
    }
}

// ---------------------------------------------------------------------------
// Token shift: dxprev = shift(x) - x ; xxx = bf16(x + dxprev * time_maa_x)
// ---------------------------------------------------------------------------
__global__ void token_shift_kernel(const float* __restrict__ x,
                                   const float* __restrict__ shift_in,
                                   const float* __restrict__ tmaax,
                                   float* __restrict__ dxprev,
                                   unsigned short* __restrict__ xxx_bf) {
    size_t idx = (size_t)blockIdx.x * blockDim.x + threadIdx.x;
    if (idx >= (size_t)M_ * C_) return;
    int c = (int)(idx % C_);
    size_t m = idx / C_;
    int t = (int)(m % T_);
    int b = (int)(m / T_);
    float xv = x[idx];
    float prev = (t == 0) ? shift_in[(size_t)b * C_ + c] : x[idx - C_];
    float dx = prev - xv;
    dxprev[idx] = dx;
    xxx_bf[idx] = f32_to_bf16(xv + dx * tmaax[c]);
}

// ---------------------------------------------------------------------------
// xq = bf16(x + dxprev * (tmaa + mix))
// ---------------------------------------------------------------------------
__global__ void mix_apply_kernel(const float* __restrict__ x,
                                 const float* __restrict__ dxprev,
                                 const float* __restrict__ tmaa,
                                 const float* __restrict__ mix,
                                 unsigned short* __restrict__ out) {
    size_t idx = (size_t)blockIdx.x * blockDim.x + threadIdx.x;
    if (idx >= (size_t)M_ * C_) return;
    int c = (int)(idx % C_);
    out[idx] = f32_to_bf16(x[idx] + dxprev[idx] * (tmaa[c] + mix[idx]));
}

// ---------------------------------------------------------------------------
// w = td + wlora; decay = exp(-exp(w)); k *= (1-decay); v2 += v2b
// ---------------------------------------------------------------------------
__global__ void decay_fuse_kernel(float* __restrict__ k,
                                  const float* __restrict__ wlora,
                                  const float* __restrict__ tdec,
                                  float* __restrict__ decay,
                                  float* __restrict__ v2,
                                  const float* __restrict__ v2b) {
    size_t idx = (size_t)blockIdx.x * blockDim.x + threadIdx.x;
    if (idx >= (size_t)M_ * C_) return;
    int c = (int)(idx % C_);
    float w = tdec[c] + wlora[idx];
    float dec = __expf(-__expf(w));
    decay[idx] = dec;
    k[idx] = k[idx] * (1.0f - dec);
    v2[idx] += v2b[idx];
}

// ---------------------------------------------------------------------------
// WKV recurrence: one block per (b,h), 64 lanes, state[64] in registers.
// ---------------------------------------------------------------------------
__global__ __launch_bounds__(64) void wkv_scan_kernel(
        const float* __restrict__ r, const float* __restrict__ k,
        const float* __restrict__ v, const float* __restrict__ d,
        const float* __restrict__ state_in,
        float* __restrict__ y, float* __restrict__ state_out) {
    int bh = blockIdx.x;
    int b  = bh / H_;
    int j  = threadIdx.x;                 // v-dim lane 0..63
    float st[N_];
    const float* si = state_in + (size_t)bh * N_ * N_;
#pragma unroll
    for (int i = 0; i < N_; ++i) st[i] = si[(size_t)i * N_ + j];

    __shared__ float sr[N_], sk[N_], sd[N_];
    size_t base = (size_t)b * T_ * C_ + (size_t)(bh % H_) * N_;
    for (int t = 0; t < T_; ++t) {
        size_t off = base + (size_t)t * C_;
        sr[j] = r[off + j];
        sk[j] = k[off + j];
        sd[j] = d[off + j];
        float vj = v[off + j];
        __syncthreads();
        float acc = 0.0f;
#pragma unroll
        for (int i = 0; i < N_; ++i) {
            acc   = fmaf(sr[i], st[i], acc);
            st[i] = fmaf(st[i], sd[i], sk[i] * vj);
        }
        y[off + j] = acc;
        __syncthreads();
    }
    float* so = state_out + (size_t)bh * N_ * N_;
#pragma unroll
    for (int i = 0; i < N_; ++i) so[(size_t)i * N_ + j] = st[i];
}

// ---------------------------------------------------------------------------
// y = LN(y + v2) -> bf16
// ---------------------------------------------------------------------------
__global__ __launch_bounds__(256) void add_ln_kernel(
        const float* __restrict__ y, const float* __restrict__ v2,
        const float* __restrict__ lnw, const float* __restrict__ lnb,
        unsigned short* __restrict__ out) {
    int row = blockIdx.x, tid = threadIdx.x;
    const float* yr = y  + (size_t)row * C_;
    const float* vr = v2 + (size_t)row * C_;
    __shared__ float buf[C_];
    __shared__ float red[256], red2[256];
    float s = 0.0f, s2 = 0.0f;
    for (int c = tid; c < C_; c += 256) {
        float val = yr[c] + vr[c];
        buf[c] = val;
        s += val; s2 += val * val;
    }
    red[tid] = s; red2[tid] = s2;
    __syncthreads();
    for (int stp = 128; stp > 0; stp >>= 1) {
        if (tid < stp) { red[tid] += red[tid + stp]; red2[tid] += red2[tid + stp]; }
        __syncthreads();
    }
    float mu  = red[0] * (1.0f / C_);
    float var = red2[0] * (1.0f / C_) - mu * mu;
    float rstd = rsqrtf(var + 1e-5f);
    for (int c = tid; c < C_; c += 256)
        out[(size_t)row * C_ + c] = f32_to_bf16((buf[c] - mu) * rstd * lnw[c] + lnb[c]);
}

__global__ void last_token_kernel(const float* __restrict__ x, float* __restrict__ out) {
    int idx = blockIdx.x * blockDim.x + threadIdx.x;
    if (idx >= B_ * C_) return;
    int b = idx / C_, c = idx % C_;
    out[idx] = x[((size_t)b * T_ + (T_ - 1)) * (size_t)C_ + c];
}

// ---------------------------------------------------------------------------
// bf16 WMMA GEMM:  D[M,N] = act(A[M,K] @ Bt[N,K]^T)
// Block tile 128x128, BK=32, 256 threads (8 waves, 4x2 wave grid); each wave
// computes 32x64 via 2x4 V_WMMA_F32_16X16X32_BF16. Tiles stream into LDS with
// GLOBAL_LOAD_ASYNC_TO_LDS_B128 into double buffers (ASYNCcnt-tracked); the
// async engine fills tile kt+1 while the WMMA pipe consumes tile kt.
// Fragment packing per CDNA5 ISA §7.12.2.
// ---------------------------------------------------------------------------
template <bool OUT_BF16, bool TANH_ACT>
__global__ __launch_bounds__(256) void wmma_gemm_kernel(
        const unsigned short* __restrict__ A, int lda,
        const unsigned short* __restrict__ Bt, int ldb,
        void* __restrict__ Dv, int ldc,
        int M, int N, int K) {
    constexpr int BM = 128, BN = 128, BK = 32;
    constexpr int LDSW = BK / 2 + 4;            // 20 dwords/row: padded, 16B-aligned
    __shared__ unsigned int ldsA[2][BM * LDSW];
    __shared__ unsigned int ldsB[2][BN * LDSW];

    const int tid  = threadIdx.x;
    const int wave = tid >> 5;
    const int lane = tid & 31;
    const int half = lane >> 4;
    const int l16  = lane & 15;
    const int wr   = wave >> 1, wc = wave & 1;  // 4x2 wave grid -> 32x64 per wave
    const int row0 = blockIdx.y * BM;
    const int col0 = blockIdx.x * BN;

    // per-thread staging chunk coordinates (2 chunks of 16B per tile)
    const int r0c = tid >> 2,            c0c = (tid & 3) * 4;
    const int r1c = (tid + 256) >> 2,    c1c = ((tid + 256) & 3) * 4;

    // zero-fill LDS rows for out-of-range B columns (done once; async issues
    // below are lane-masked for those rows and never overwrite them)
    {
        uint4 z = {0u, 0u, 0u, 0u};
        if (col0 + r0c >= N) {
            *(uint4*)&ldsB[0][r0c * LDSW + c0c] = z;
            *(uint4*)&ldsB[1][r0c * LDSW + c0c] = z;
        }
        if (col0 + r1c >= N) {
            *(uint4*)&ldsB[0][r1c * LDSW + c1c] = z;
            *(uint4*)&ldsB[1][r1c * LDSW + c1c] = z;
        }
    }
    __syncthreads();

    union CF { v8f v; float f[8]; };
    CF acc[2][4];
#pragma unroll
    for (int a = 0; a < 2; ++a)
#pragma unroll
        for (int b = 0; b < 4; ++b)
#pragma unroll
            for (int i = 0; i < 8; ++i) acc[a][b].f[i] = 0.0f;

    const int nK = K / BK;

    // helper to issue one tile's async loads into buffer `buf`
    auto issue_tile = [&](int buf, int k0) {
        async_load_b128(A + (size_t)(row0 + r0c) * lda + k0 + c0c * 2,
                        &ldsA[buf][r0c * LDSW + c0c]);
        async_load_b128(A + (size_t)(row0 + r1c) * lda + k0 + c1c * 2,
                        &ldsA[buf][r1c * LDSW + c1c]);
        if (col0 + r0c < N)
            async_load_b128(Bt + (size_t)(col0 + r0c) * ldb + k0 + c0c * 2,
                            &ldsB[buf][r0c * LDSW + c0c]);
        if (col0 + r1c < N)
            async_load_b128(Bt + (size_t)(col0 + r1c) * ldb + k0 + c1c * 2,
                            &ldsB[buf][r1c * LDSW + c1c]);
    };

    issue_tile(0, 0);
    for (int kt = 0; kt < nK; ++kt) {
        const int cur = kt & 1;
        wait_async0();          // this wave's async copies for buffer `cur` done
        __syncthreads();        // ... and everyone else's; prev reads finished
        if (kt + 1 < nK) {
            issue_tile(1 - cur, (kt + 1) * BK);
            if (kt + 2 < nK) {  // speculative L2 prefetch of the tile after next
                __builtin_prefetch(A + (size_t)(row0 + (tid & 127)) * lda + (kt + 2) * BK, 0, 1);
                __builtin_prefetch(Bt + (size_t)(col0 + (tid & 127)) * ldb + (kt + 2) * BK, 0, 1);
            }
        }

        union BFr { v16bf v; unsigned int u[8]; };
        BFr af[2], bf[4];
        // A fragment: lane half h holds K = {8h..8h+7, 16+8h..16+8h+7} packed pairs
#pragma unroll
        for (int rt = 0; rt < 2; ++rt) {
            int arow = wr * 32 + rt * 16 + l16;
            uint4 lo = *(const uint4*)&ldsA[cur][arow * LDSW + 4 * half];
            uint4 hi = *(const uint4*)&ldsA[cur][arow * LDSW + 8 + 4 * half];
            af[rt].u[0] = lo.x; af[rt].u[1] = lo.y; af[rt].u[2] = lo.z; af[rt].u[3] = lo.w;
            af[rt].u[4] = hi.x; af[rt].u[5] = hi.y; af[rt].u[6] = hi.z; af[rt].u[7] = hi.w;
        }
        // B fragment: lane half h holds K = {8h..8h+7, 16+8h..16+8h+7} packed pairs for column l16
#pragma unroll
        for (int ct = 0; ct < 4; ++ct) {
            int bn = wc * 64 + ct * 16 + l16;
            uint4 lo = *(const uint4*)&ldsB[cur][bn * LDSW + 4 * half];
            uint4 hi = *(const uint4*)&ldsB[cur][bn * LDSW + 8 + 4 * half];
            bf[ct].u[0] = lo.x; bf[ct].u[1] = lo.y; bf[ct].u[2] = lo.z; bf[ct].u[3] = lo.w;
            bf[ct].u[4] = hi.x; bf[ct].u[5] = hi.y; bf[ct].u[6] = hi.z; bf[ct].u[7] = hi.w;
        }
#pragma unroll
        for (int rt = 0; rt < 2; ++rt)
#pragma unroll
            for (int ct = 0; ct < 4; ++ct)
                acc[rt][ct].v = __builtin_amdgcn_wmma_f32_16x16x32_bf16(
                    false, af[rt].v, false, bf[ct].v,
                    (short)0, acc[rt][ct].v, false, false);
        // next overwrite of buffer `cur` is gated by the barrier in iteration
        // kt+1, which all waves reach only after finishing these reads.
    }

    // ---- store D: VGPR v -> M = v + 8*half, N = l16 ----
#pragma unroll
    for (int rt = 0; rt < 2; ++rt) {
#pragma unroll
        for (int ct = 0; ct < 4; ++ct) {
            int bn = col0 + wc * 64 + ct * 16 + l16;
            if (bn < N) {
#pragma unroll
                for (int vv = 0; vv < 8; ++vv) {
                    int gm = row0 + wr * 32 + rt * 16 + vv + 8 * half;
                    float val = acc[rt][ct].f[vv];
                    if (TANH_ACT) val = tanhf(val);
                    if (OUT_BF16)
                        ((unsigned short*)Dv)[(size_t)gm * ldc + bn] = f32_to_bf16(val);
                    else
                        ((float*)Dv)[(size_t)gm * ldc + bn] = val;
                }
            }
        }
    }
}

// ---------------------------------------------------------------------------
static void launch_gemm(const unsigned short* A, int lda,
                        const unsigned short* Bt, int ldb,
                        void* D, int ldc, int M, int N, int K,
                        bool out_bf16_tanh, hipStream_t s) {
    dim3 grid((N + 127) / 128, M / 128), block(256);
    if (out_bf16_tanh)
        wmma_gemm_kernel<true, true><<<grid, block, 0, s>>>(A, lda, Bt, ldb, D, ldc, M, N, K);
    else
        wmma_gemm_kernel<false, false><<<grid, block, 0, s>>>(A, lda, Bt, ldb, D, ldc, M, N, K);
}

static void launch_cvt_t(const float* in, unsigned short* out, int R, int Cc, hipStream_t s) {
    long total = (long)R * Cc;
    cvt_transpose_bf16_kernel<<<(unsigned)((total + 255) / 256), 256, 0, s>>>(in, out, R, Cc);
}

extern "C" void kernel_launch(void* const* d_in, const int* in_sizes, int n_in,
                              void* d_out, int out_size, void* d_ws, size_t ws_size,
                              hipStream_t stream) {
    (void)in_sizes; (void)n_in; (void)out_size; (void)ws_size;
    const float* x      = (const float*)d_in[0];
    const float* shift  = (const float*)d_in[1];
    const float* wkv_in = (const float*)d_in[2];
    const float* tmaax  = (const float*)d_in[3];
    const float* tmaa_q[5] = {(const float*)d_in[4], (const float*)d_in[5],
                              (const float*)d_in[6], (const float*)d_in[7],
                              (const float*)d_in[8]};
    const float* w1   = (const float*)d_in[9];
    const float* w2   = (const float*)d_in[10];
    const float* tdec = (const float*)d_in[11];
    const float* tdw1 = (const float*)d_in[12];
    const float* tdw2 = (const float*)d_in[13];
    const float* tvw1 = (const float*)d_in[14];
    const float* tvw2 = (const float*)d_in[15];
    const float* Wr = (const float*)d_in[16];
    const float* Wk = (const float*)d_in[17];
    const float* Wv = (const float*)d_in[18];
    const float* Wo = (const float*)d_in[19];
    const float* lnw = (const float*)d_in[20];
    const float* lnb = (const float*)d_in[21];
    float* out = (float*)d_out;

    char* wsb = (char*)d_ws;
    size_t off = 0;
    auto alloc = [&](size_t bytes) -> void* {
        void* p = wsb + off; off += (bytes + 255) & ~(size_t)255; return p;
    };
    const size_t MC = (size_t)M_ * C_;

    float*          dxprev = (float*)alloc(MC * 4);           // reused later as v2b
    unsigned short* xxx_bf = (unsigned short*)alloc(MC * 2);
    unsigned short* xq_bf[5];
    for (int f = 0; f < 5; ++f) xq_bf[f] = (unsigned short*)alloc(MC * 2);
    unsigned short* hmix = (unsigned short*)alloc((size_t)M_ * 5 * DMIX_ * 2);
    float*          mixbuf = (float*)alloc(MC * 4);           // reused later as wlora
    unsigned short* WrT = (unsigned short*)alloc((size_t)C_ * C_ * 2);
    unsigned short* WkT = (unsigned short*)alloc((size_t)C_ * C_ * 2);
    unsigned short* WvT = (unsigned short*)alloc((size_t)C_ * C_ * 2);
    unsigned short* WoT = (unsigned short*)alloc((size_t)C_ * C_ * 2);
    unsigned short* w1t = (unsigned short*)alloc((size_t)5 * DMIX_ * C_ * 2);
    unsigned short* w2t = (unsigned short*)alloc((size_t)5 * C_ * DMIX_ * 2);
    unsigned short* tdw1t = (unsigned short*)alloc((size_t)DDEC_ * C_ * 2);
    unsigned short* tdw2t = (unsigned short*)alloc((size_t)C_ * DDEC_ * 2);
    unsigned short* tvw1t = (unsigned short*)alloc((size_t)DDEC_ * C_ * 2);
    unsigned short* tvw2t = (unsigned short*)alloc((size_t)C_ * DDEC_ * 2);
    float* rbuf   = (float*)alloc(MC * 4);
    float* kbuf   = (float*)alloc(MC * 4);
    float* vbuf   = (float*)alloc(MC * 4);
    float* decbuf = (float*)alloc(MC * 4);
    float* v2buf  = (float*)alloc(MC * 4);
    float* ybuf   = (float*)alloc(MC * 4);
    unsigned short* hdec = (unsigned short*)alloc((size_t)M_ * DDEC_ * 2);
    unsigned short* hv2  = (unsigned short*)alloc((size_t)M_ * DDEC_ * 2);
    unsigned short* yln  = (unsigned short*)alloc(MC * 2);

    // ---- 1. weight conversions (f32 -> bf16, transposed to [N,K]) ----
    launch_cvt_t(w1, w1t, C_, 5 * DMIX_, stream);
    for (int f = 0; f < 5; ++f)
        launch_cvt_t(w2 + (size_t)f * DMIX_ * C_, w2t + (size_t)f * C_ * DMIX_,
                     DMIX_, C_, stream);
    launch_cvt_t(Wr, WrT, C_, C_, stream);
    launch_cvt_t(Wk, WkT, C_, C_, stream);
    launch_cvt_t(Wv, WvT, C_, C_, stream);
    launch_cvt_t(Wo, WoT, C_, C_, stream);
    launch_cvt_t(tdw1, tdw1t, C_, DDEC_, stream);
    launch_cvt_t(tdw2, tdw2t, DDEC_, C_, stream);
    launch_cvt_t(tvw1, tvw1t, C_, DDEC_, stream);
    launch_cvt_t(tvw2, tvw2t, DDEC_, C_, stream);

    // ---- 2. token shift ----
    unsigned ew_blocks = (unsigned)((MC + 255) / 256);
    token_shift_kernel<<<ew_blocks, 256, 0, stream>>>(x, shift, tmaax, dxprev, xxx_bf);

    // ---- 3. hmix = tanh(xxx @ w1)  [M, 160] bf16 ----
    launch_gemm(xxx_bf, C_, w1t, C_, hmix, 5 * DMIX_, M_, 5 * DMIX_, C_, true, stream);

    // ---- 4. five K=32 mix GEMMs + apply ----
    for (int f = 0; f < 5; ++f) {
        launch_gemm(hmix + f * DMIX_, 5 * DMIX_, w2t + (size_t)f * C_ * DMIX_, DMIX_,
                    mixbuf, C_, M_, C_, DMIX_, false, stream);
        mix_apply_kernel<<<ew_blocks, 256, 0, stream>>>(x, dxprev, tmaa_q[f], mixbuf,
                                                        xq_bf[f]);
    }

    // ---- 5. projections ----
    launch_gemm(xq_bf[0], C_, WrT, C_, rbuf,  C_, M_, C_, C_, false, stream);  // r
    launch_gemm(xq_bf[1], C_, WkT, C_, kbuf,  C_, M_, C_, C_, false, stream);  // k
    launch_gemm(xq_bf[2], C_, WvT, C_, vbuf,  C_, M_, C_, C_, false, stream);  // v
    launch_gemm(xq_bf[4], C_, WvT, C_, v2buf, C_, M_, C_, C_, false, stream);  // v2a
    // decay LoRA: hdec = tanh(xw @ tdw1); wlora = hdec @ tdw2 (reuse mixbuf)
    launch_gemm(xq_bf[3], C_, tdw1t, C_, hdec, DDEC_, M_, DDEC_, C_, true, stream);
    launch_gemm(hdec, DDEC_, tdw2t, DDEC_, mixbuf, C_, M_, C_, DDEC_, false, stream);
    // v2 LoRA: hv2 = tanh(xv2 @ tvw1); v2b = hv2 @ tvw2 (reuse dxprev)
    launch_gemm(xq_bf[4], C_, tvw1t, C_, hv2, DDEC_, M_, DDEC_, C_, true, stream);
    launch_gemm(hv2, DDEC_, tvw2t, DDEC_, dxprev, C_, M_, C_, DDEC_, false, stream);

    // ---- 6. decay/k/v2 fuse ----
    decay_fuse_kernel<<<ew_blocks, 256, 0, stream>>>(kbuf, mixbuf, tdec, decbuf,
                                                     v2buf, dxprev);

    // ---- 7. WKV scan (writes final state directly into d_out) ----
    float* out_state = out + MC + (size_t)B_ * C_;
    wkv_scan_kernel<<<B_ * H_, N_, 0, stream>>>(rbuf, kbuf, vbuf, decbuf, wkv_in,
                                                ybuf, out_state);

    // ---- 8. x[:, -1] ----
    last_token_kernel<<<(B_ * C_ + 255) / 256, 256, 0, stream>>>(x, out + MC);

    // ---- 9. y = LN(y + v2) -> bf16 ----
    add_ln_kernel<<<M_, 256, 0, stream>>>(ybuf, v2buf, lnw, lnb, yln);

    // ---- 10. out = yln @ W_o -> d_out ----
    launch_gemm(yln, C_, WoT, C_, out, C_, M_, C_, C_, false, stream);
}
